// SparseAttention_87273735455080
// MI455X (gfx1250) — matrix operand order repaired and checked by reading, not compiled, not run on hardware
//
#include <hip/hip_runtime.h>
#include <hip/hip_bf16.h>

// ---------------------------------------------------------------------------
// Longformer sparse attention, MI455X (gfx1250, wave32, WMMA + TDM).
// f32 inputs converted once to bf16; all GEMMs + attention use
// v_wmma_f32_16x16x32_bf16 (f32 accumulate). GEMM block tiles are staged into
// LDS by the Tensor Data Mover (double-buffered, TENSORcnt-synchronized),
// using TDM pad insertion (80B row stride) for conflict-free ds_load_b128.
// ---------------------------------------------------------------------------

typedef __attribute__((ext_vector_type(16))) __bf16 v16bf;
typedef __attribute__((ext_vector_type(8)))  __bf16 v8bf;
typedef __attribute__((ext_vector_type(4)))  __bf16 v4bf;
typedef __attribute__((ext_vector_type(8)))  float  v8f;
typedef __attribute__((ext_vector_type(4)))  unsigned int v4u;
typedef __attribute__((ext_vector_type(8)))  int    v8i;
typedef __attribute__((ext_vector_type(4)))  int    v4i;

static constexpr int Bsz    = 2;
static constexpr int Sseq   = 2048;
static constexpr int HIDdim = 1024;
static constexpr int NHn    = 16;
static constexpr int HDd    = 64;
static constexpr int MROWS  = Bsz * Sseq;      // 4096
static constexpr int WHALF  = 64;              // LOCAL_WINDOW / 2

#if defined(__has_builtin)
#  if __has_builtin(__builtin_amdgcn_tensor_load_to_lds)
#    define HAVE_TDM 1
#  else
#    define HAVE_TDM 0
#  endif
#else
#  define HAVE_TDM 0
#endif

// round-to-nearest-even f32 -> bf16
__device__ __forceinline__ __bf16 f2bf(float f) {
    unsigned u = __builtin_bit_cast(unsigned, f);
    u += 0x7FFFu + ((u >> 16) & 1u);
    unsigned short h = (unsigned short)(u >> 16);
    return __builtin_bit_cast(__bf16, h);
}

__device__ __forceinline__ v16bf cat8(v8bf lo, v8bf hi) {
    return __builtin_shufflevector(lo, hi, 0,1,2,3,4,5,6,7,8,9,10,11,12,13,14,15);
}

// ---------------------------------------------------------------------------
// f32 -> bf16 bulk convert (x4 vectorized); ~25MB once, ~1us at HBM peak.
// ---------------------------------------------------------------------------
__global__ __launch_bounds__(256) void cvt_f32_bf16(const float* __restrict__ src,
                                                    __bf16* __restrict__ dst, int n4) {
    int i = blockIdx.x * blockDim.x + threadIdx.x;
    int stride = gridDim.x * blockDim.x;
    for (; i < n4; i += stride) {
        float4 f = reinterpret_cast<const float4*>(src)[i];
        v4bf o;
        o[0] = f2bf(f.x); o[1] = f2bf(f.y); o[2] = f2bf(f.z); o[3] = f2bf(f.w);
        reinterpret_cast<v4bf*>(dst)[i] = o;
    }
}

// ---------------------------------------------------------------------------
// TDM tile descriptor: 2D tile (tile_d1 rows x 32 bf16) from a row-major
// bf16 tensor with row stride `stride0` elems, into LDS at `lds_addr`.
// pad_enable: +16B LDS padding after every 64B row -> 80B LDS row stride.
// ---------------------------------------------------------------------------
static constexpr int KSTEP = 32;
static constexpr int ROWB  = 80;                 // padded LDS row stride (bytes)
static constexpr int XROWS = 64, WROWS = 128;    // block tile: 64 x 128
static constexpr int XSZ   = XROWS * ROWB;       // 5120 B
static constexpr int WSZ   = WROWS * ROWB;       // 10240 B
static constexpr int BUFSZ = XSZ + WSZ;          // 15360 B (x2 buffers = 30 KB)

#if HAVE_TDM
__device__ __forceinline__ void tdm_load_tile(unsigned lds_addr, unsigned long long gaddr,
                                              unsigned tensor_d0, unsigned tensor_d1,
                                              unsigned tile_d1, unsigned stride0) {
    v4u g0;
    g0[0] = 1u;                                          // count=1 (valid), user mode
    g0[1] = lds_addr;                                    // LDS byte address
    g0[2] = (unsigned)(gaddr & 0xFFFFFFFFull);           // global_addr[31:0]
    g0[3] = (unsigned)((gaddr >> 32) & 0x01FFFFFFull) | (2u << 30);  // [56:32] | type=2
    v8i g1;
    g1[0] = (int)((1u << 16) |                           // data_size = 2 bytes
                  (1u << 20) |                           // pad_enable
                  (3u << 22) |                           // pad_interval: 16 DWORDs (64B)
                  (3u << 25));                           // pad_amount: 4 DWORDs (16B)
    g1[1] = (int)((tensor_d0 & 0xFFFFu) << 16);          // abar_addr=0 | tensor_dim0 lo
    g1[2] = (int)((tensor_d0 >> 16) | ((tensor_d1 & 0xFFFFu) << 16));
    g1[3] = (int)((tensor_d1 >> 16) | ((unsigned)KSTEP << 16));  // | tile_dim0 = 32
    g1[4] = (int)(tile_d1 & 0xFFFFu);                    // tile_dim1 | tile_dim2 = 0
    g1[5] = (int)stride0;                                // tensor_dim0_stride[31:0]
    g1[6] = 0;                                           // stride0[47:32] | stride1 lo
    g1[7] = 0;
    v4i z = {0, 0, 0, 0};
#if __clang_major__ >= 23
    v8i z8 = {0, 0, 0, 0, 0, 0, 0, 0};
    __builtin_amdgcn_tensor_load_to_lds(g0, g1, z, z, z8, 0);
#else
    __builtin_amdgcn_tensor_load_to_lds(g0, g1, z, z, 0);
#endif
}
#else
// Fallback: cooperative global->LDS copy replicating the padded layout.
__device__ __forceinline__ void coop_copy_tile(unsigned char* dst, const __bf16* src,
                                               int rows, int Kdim, int tid) {
    for (int idx = tid; idx < rows * 4; idx += 256) {
        const int r = idx >> 2, c = idx & 3;
        v8bf v = *reinterpret_cast<const v8bf*>(src + (size_t)r * Kdim + c * 8);
        *reinterpret_cast<v8bf*>(dst + r * ROWB + c * 16) = v;
    }
}
#endif

// ---------------------------------------------------------------------------
// Y[M,N] = Xbf16[M,K] @ Wbf16[N,K]^T + bias.
// Block = 8 waves (2x4) = 64x128 tile; wave = 32x32 (2x2 WMMA accumulators).
// TDM stages X(64x32) + W(128x32) bf16 panels per k-step, double-buffered.
// out_mode: 0 -> bf16 out [B,NH,S,HD] (Q,K); 1 -> bf16 out [B,NH,HD,S] (V^T);
//           2 -> f32 out row-major [M,N] (final projection).
// ---------------------------------------------------------------------------
__global__ __launch_bounds__(256)
void gemm_bf16_wmma(const __bf16* __restrict__ X, const __bf16* __restrict__ W,
                    const float* __restrict__ bias, __bf16* __restrict__ Yb,
                    float* __restrict__ Yf, int Mrows, int Ncols, int Kdim,
                    int out_mode) {
    __shared__ __align__(16) unsigned char smem[2 * BUFSZ];

    const int tid  = threadIdx.x;
    const int lane = tid & 31;
    const int wid  = tid >> 5;
    const int half = lane >> 4, l16 = lane & 15;

    const int nblk = Ncols / 128;
    const int bm = blockIdx.x / nblk, bn = blockIdx.x % nblk;
    const int m0 = bm * 64, n0 = bn * 128;
    const int mw = (wid >> 2) * 32, nw = (wid & 3) * 32;   // wave offsets in block tile
    const int nsteps = Kdim / KSTEP;

    v8f acc[2][2] = {};

#if HAVE_TDM
    const unsigned ldsbase = (unsigned)(size_t)(&smem[0]);
    const unsigned long long xg0 = (unsigned long long)(size_t)X + (size_t)m0 * Kdim * 2;
    const unsigned long long wg0 = (unsigned long long)(size_t)W + (size_t)n0 * Kdim * 2;
    if (wid == 0) {                                        // wave-uniform
        tdm_load_tile(ldsbase,        xg0, Kdim, Mrows, XROWS, Kdim);
        tdm_load_tile(ldsbase + XSZ,  wg0, Kdim, Ncols, WROWS, Kdim);
    }
#else
    coop_copy_tile(smem,       X + (size_t)m0 * Kdim, XROWS, Kdim, tid);
    coop_copy_tile(smem + XSZ, W + (size_t)n0 * Kdim, WROWS, Kdim, tid);
    __syncthreads();
#endif

    for (int s = 0; s < nsteps; ++s) {
        const int buf = s & 1;
#if HAVE_TDM
        if (wid == 0) {
            if (s + 1 < nsteps) {
                const unsigned long long koff = (unsigned long long)(s + 1) * KSTEP * 2;
                const unsigned lb = ldsbase + (buf ^ 1) * BUFSZ;
                tdm_load_tile(lb,       xg0 + koff, Kdim, Mrows, XROWS, Kdim);
                tdm_load_tile(lb + XSZ, wg0 + koff, Kdim, Ncols, WROWS, Kdim);
                __builtin_amdgcn_s_wait_tensorcnt(2);      // oldest pair (this buf) done
            } else {
                __builtin_amdgcn_s_wait_tensorcnt(0);
            }
        }
        __syncthreads();                                   // buf ready for all waves
#else
        if (s + 1 < nsteps) {
            const int koff = (s + 1) * KSTEP;
            unsigned char* lb = smem + (buf ^ 1) * BUFSZ;
            coop_copy_tile(lb,       X + (size_t)m0 * Kdim + koff, XROWS, Kdim, tid);
            coop_copy_tile(lb + XSZ, W + (size_t)n0 * Kdim + koff, WROWS, Kdim, tid);
        }
#endif
        const unsigned char* xs = smem + buf * BUFSZ;
        const unsigned char* ws = xs + XSZ;

        v16bf a[2], bt[2];
#pragma unroll
        for (int i = 0; i < 2; ++i) {
            // A operand: lane = row (l16), contiguous K-chunk at half*16 elems
            const unsigned char* pa = xs + (mw + i * 16 + l16) * ROWB + half * 32;
            v8bf alo = *reinterpret_cast<const v8bf*>(pa);
            v8bf ahi = *reinterpret_cast<const v8bf*>(pa + 16);
            a[i] = cat8(alo, ahi);
            // B operand: lane = column (l16), contiguous K-chunk at half*16 elems
            const unsigned char* pb = ws + (nw + i * 16 + l16) * ROWB + half * 32;
            v8bf blo = *reinterpret_cast<const v8bf*>(pb);
            v8bf bhi = *reinterpret_cast<const v8bf*>(pb + 16);
            bt[i] = cat8(blo, bhi);
        }
#pragma unroll
        for (int i = 0; i < 2; ++i)
#pragma unroll
            for (int j = 0; j < 2; ++j)
                acc[i][j] = __builtin_amdgcn_wmma_f32_16x16x32_bf16(
                    false, a[i], false, bt[j], (short)0, acc[i][j], false, false);

        __syncthreads();                                   // reads done before overwrite
    }

    // Epilogue: D layout = row (r + 8*half) x col (l16)
#pragma unroll
    for (int i = 0; i < 2; ++i) {
        const int row_base = m0 + mw + i * 16 + 8 * half;
#pragma unroll
        for (int j = 0; j < 2; ++j) {
            const int col = n0 + nw + j * 16 + l16;
            const float bv = bias[col];
#pragma unroll
            for (int r = 0; r < 8; ++r) {
                const int row = row_base + r;
                const float val = acc[i][j][r] + bv;
                if (out_mode == 2) {
                    Yf[(size_t)row * Ncols + col] = val;
                } else {
                    const int b = row >> 11, sidx = row & (Sseq - 1);
                    const int h = col >> 6,  d = col & (HDd - 1);
                    size_t idx;
                    if (out_mode == 0)
                        idx = (((size_t)(b * NHn + h)) * Sseq + sidx) * HDd + d;
                    else
                        idx = (((size_t)(b * NHn + h)) * HDd + d) * Sseq + sidx;
                    Yb[idx] = f2bf(val);
                }
            }
        }
    }
}

// ---------------------------------------------------------------------------
// Sparse attention. One wave per (b, h, 16-row query tile). Key tiles:
//   s = 0..4 : t_start + s, t_start = max(qt-4, 0)  (local window, causal)
//   s = 5    : tile 0 (global column), active only when t_start > 0
// All 6 score tiles in registers; softmax via 16-lane xor-shuffles; P staged
// through LDS as bf16 to re-layout into the WMMA A-operand.
// ---------------------------------------------------------------------------
__global__ __launch_bounds__(256)
void attn_sparse_wmma(const __bf16* __restrict__ Q, const __bf16* __restrict__ K,
                      const __bf16* __restrict__ Vt, __bf16* __restrict__ Aout) {
    __shared__ __align__(16) __bf16 plds[8][16][104];   // 96 keys + pad (16B-mult)

    const int lane = threadIdx.x & 31;
    const int wid  = threadIdx.x >> 5;
    const int gw   = blockIdx.x * 8 + wid;
    const int QT   = Sseq / 16;                  // 128
    const int qt   = gw % QT;
    const int bh   = gw / QT;
    const int b    = bh / NHn, h = bh % NHn;
    const int q0   = qt * 16;
    const int half = lane >> 4, l16 = lane & 15;

    const __bf16* Qp = Q  + (size_t)bh * Sseq * HDd;
    const __bf16* Kp = K  + (size_t)bh * Sseq * HDd;
    const __bf16* Vp = Vt + (size_t)bh * HDd * Sseq;

    // Q A-operands for the two d-steps (d0 = 0, 32)
    v16bf qa[2];
    {
        const __bf16* qrow = Qp + (size_t)(q0 + l16) * HDd + half * 16;
#pragma unroll
        for (int ds = 0; ds < 2; ++ds) {
            v8bf lo = *reinterpret_cast<const v8bf*>(qrow + ds * 32);
            v8bf hi = *reinterpret_cast<const v8bf*>(qrow + ds * 32 + 8);
            qa[ds] = cat8(lo, hi);
        }
    }

    const int t_start = (qt > 4) ? (qt - 4) : 0;
    const bool g_ok = (t_start > 0);
    int kt[6];
#pragma unroll
    for (int s = 0; s < 5; ++s) kt[s] = t_start + s;
    kt[5] = 0;

    // S = Q K^T for 6 key tiles (2 WMMAs each over d=64)
    v8f sc[6];
#pragma unroll
    for (int s = 0; s < 6; ++s) {
        const __bf16* krow = Kp + (size_t)(kt[s] * 16 + l16) * HDd + half * 16;
        v8f a = {};
#pragma unroll
        for (int ds = 0; ds < 2; ++ds) {
            v8bf lo = *reinterpret_cast<const v8bf*>(krow + ds * 32);
            v8bf hi = *reinterpret_cast<const v8bf*>(krow + ds * 32 + 8);
            v16bf kb = cat8(lo, hi);
            a = __builtin_amdgcn_wmma_f32_16x16x32_bf16(
                false, qa[ds], false, kb, (short)0, a, false, false);
        }
        sc[s] = a;
    }

    // mask (causal + window + global row/col) and scale by 1/sqrt(64)
    const float scale = 0.125f;
#pragma unroll
    for (int s = 0; s < 6; ++s) {
        const int j = kt[s] * 16 + l16;
        const bool tile_ok = (s < 5) || g_ok;
#pragma unroll
        for (int r = 0; r < 8; ++r) {
            const int i = q0 + r + 8 * half;
            const bool ok = tile_ok && (j <= i) &&
                            ((i - j) <= WHALF || i == 0 || j == 0);
            sc[s][r] = ok ? sc[s][r] * scale : -1e30f;
        }
    }

    // softmax: row max / exp / row sum (rows live across 16-lane groups)
    v8f rm, rs;
#pragma unroll
    for (int r = 0; r < 8; ++r) {
        float m = sc[0][r];
#pragma unroll
        for (int s = 1; s < 6; ++s) m = fmaxf(m, sc[s][r]);
#pragma unroll
        for (int off = 8; off >= 1; off >>= 1)
            m = fmaxf(m, __shfl_xor(m, off, 32));
        rm[r] = m;
    }
#pragma unroll
    for (int r = 0; r < 8; ++r) {
        float sum = 0.f;
#pragma unroll
        for (int s = 0; s < 6; ++s) {
            float p = __expf(sc[s][r] - rm[r]);
            sc[s][r] = p;
            sum += p;
        }
#pragma unroll
        for (int off = 8; off >= 1; off >>= 1)
            sum += __shfl_xor(sum, off, 32);
        rs[r] = sum;
    }

    // stage unnormalized P into LDS as bf16 (row-major per wave)
#pragma unroll
    for (int s = 0; s < 6; ++s)
#pragma unroll
        for (int r = 0; r < 8; ++r)
            plds[wid][r + 8 * half][s * 16 + l16] = f2bf(sc[s][r]);
    // same-wave DS store->load: LDS ops execute in order; compiler inserts waits.

    // O = P @ V  (K = 96 keys in 3 steps of 32; N = 64 in 4 column tiles)
    v8f oacc[4] = {};
#pragma unroll
    for (int kk = 0; kk < 3; ++kk) {
        const __bf16* prow = &plds[wid][l16][kk * 32 + half * 16];
        v8bf plo = *reinterpret_cast<const v8bf*>(prow);
        v8bf phi = *reinterpret_cast<const v8bf*>(prow + 8);
        v16bf pa = cat8(plo, phi);
        const int ts   = kk * 2 + half;                        // 0..5
        const int key0 = (ts == 5) ? 0 : (t_start + ts) * 16;  // global tile base
#pragma unroll
        for (int j = 0; j < 4; ++j) {
            const __bf16* vrow = Vp + (size_t)(j * 16 + l16) * Sseq + key0;
            v8bf vlo = *reinterpret_cast<const v8bf*>(vrow);
            v8bf vhi = *reinterpret_cast<const v8bf*>(vrow + 8);
            v16bf vb = cat8(vlo, vhi);
            oacc[j] = __builtin_amdgcn_wmma_f32_16x16x32_bf16(
                false, pa, false, vb, (short)0, oacc[j], false, false);
        }
    }

    // normalize by softmax denominator and store [B, S, NH*HD] bf16
    const size_t mbase = (size_t)b * Sseq;
#pragma unroll
    for (int j = 0; j < 4; ++j) {
        const int col = h * HDd + j * 16 + l16;
#pragma unroll
        for (int r = 0; r < 8; ++r) {
            const int srow = q0 + 8 * half + r;
            Aout[(mbase + srow) * HIDdim + col] = f2bf(oacc[j][r] / rs[r]);
        }
    }
}

// ---------------------------------------------------------------------------
extern "C" void kernel_launch(void* const* d_in, const int* in_sizes, int n_in,
                              void* d_out, int out_size, void* d_ws, size_t ws_size,
                              hipStream_t stream) {
    const float* hs = (const float*)d_in[0];
    const float* Wq = (const float*)d_in[1];
    const float* bq = (const float*)d_in[2];
    const float* Wk = (const float*)d_in[3];
    const float* bk = (const float*)d_in[4];
    const float* Wv = (const float*)d_in[5];
    const float* bv = (const float*)d_in[6];
    const float* Wo = (const float*)d_in[7];
    const float* bo = (const float*)d_in[8];
    float* out = (float*)d_out;

    // workspace layout (bf16 elements): 24M elems = 48 MB
    __bf16* xb  = (__bf16*)d_ws;                          // [4096,1024] hidden
    __bf16* wqb = xb  + (size_t)MROWS * HIDdim;
    __bf16* wkb = wqb + (size_t)HIDdim * HIDdim;
    __bf16* wvb = wkb + (size_t)HIDdim * HIDdim;
    __bf16* wob = wvb + (size_t)HIDdim * HIDdim;
    __bf16* qb  = wob + (size_t)HIDdim * HIDdim;          // [B,NH,S,HD]
    __bf16* kb  = qb  + (size_t)Bsz * NHn * Sseq * HDd;   // [B,NH,S,HD]
    __bf16* vtb = kb  + (size_t)Bsz * NHn * Sseq * HDd;   // [B,NH,HD,S]
    __bf16* ab  = vtb + (size_t)Bsz * NHn * Sseq * HDd;   // [4096,1024] attn out

    cvt_f32_bf16<<<1024, 256, 0, stream>>>(hs, xb, MROWS * HIDdim / 4);
    cvt_f32_bf16<<<512, 256, 0, stream>>>(Wq, wqb, HIDdim * HIDdim / 4);
    cvt_f32_bf16<<<512, 256, 0, stream>>>(Wk, wkb, HIDdim * HIDdim / 4);
    cvt_f32_bf16<<<512, 256, 0, stream>>>(Wv, wvb, HIDdim * HIDdim / 4);
    cvt_f32_bf16<<<512, 256, 0, stream>>>(Wo, wob, HIDdim * HIDdim / 4);

    const int gblocks = (MROWS / 64) * (HIDdim / 128);      // 512 blocks (8 waves each)
    gemm_bf16_wmma<<<gblocks, 256, 0, stream>>>(xb, wqb, bq, qb, nullptr,
                                                MROWS, HIDdim, HIDdim, 0);
    gemm_bf16_wmma<<<gblocks, 256, 0, stream>>>(xb, wkb, bk, kb, nullptr,
                                                MROWS, HIDdim, HIDdim, 0);
    gemm_bf16_wmma<<<gblocks, 256, 0, stream>>>(xb, wvb, bv, vtb, nullptr,
                                                MROWS, HIDdim, HIDdim, 1);

    const int ablocks = (Bsz * NHn * (Sseq / 16)) / 8;      // 512 blocks
    attn_sparse_wmma<<<ablocks, 256, 0, stream>>>(qb, kb, vtb, ab);

    gemm_bf16_wmma<<<gblocks, 256, 0, stream>>>(ab, wob, bo, nullptr, out,
                                                MROWS, HIDdim, HIDdim, 2);
}
